// SDP_75453985457333
// MI455X (gfx1250) — matrix operand order repaired and checked by reading, not compiled, not run on hardware
//
#include <hip/hip_runtime.h>

typedef _Float16 half_t;
typedef __attribute__((ext_vector_type(16))) _Float16 v16h;
typedef __attribute__((ext_vector_type(8)))  _Float16 v8h;
typedef __attribute__((ext_vector_type(4)))  _Float16 v4h;
typedef __attribute__((ext_vector_type(8)))  float    v8f;
typedef __attribute__((ext_vector_type(4)))  float    v4f;

#define DIMC 256
#define HW   16384     // 128*128
#define EPSV 1e-5f

__device__ __forceinline__ v8f zero_v8f() {
  v8f z = {0.f,0.f,0.f,0.f,0.f,0.f,0.f,0.f};
  return z;
}

__device__ __forceinline__ v8f wmma_f16(v16h a, v16h b, v8f c) {
  return __builtin_amdgcn_wmma_f32_16x16x32_f16(false, a, false, b, (short)0, c, false, false);
}

__device__ __forceinline__ v16h cat8(v8h lo, v8h hi) {
  return __builtin_shufflevector(lo, hi, 0,1,2,3,4,5,6,7,8,9,10,11,12,13,14,15);
}

// ---- gfx1250 async global->LDS copy (ASYNCcnt path) ----
__device__ __forceinline__ unsigned lds_addr_of(const void* p) {
  // generic -> LDS address space -> 32-bit workgroup-LDS byte offset
  return (unsigned)(size_t)(__attribute__((address_space(3))) const char*)p;
}
__device__ __forceinline__ void async_copy_b128(unsigned lds_off, const void* gsrc) {
  asm volatile("global_load_async_to_lds_b128 %0, %1, off"
               :: "v"(lds_off), "v"(gsrc) : "memory");
}
__device__ __forceinline__ void wait_asynccnt0() {
  asm volatile("s_wait_asynccnt 0x0" ::: "memory");
}

// ---- WMMA fragment loaders (CDNA5 wave32 layouts, ISA 7.12.2) ----
// A (16x32) from row-major A[m][k]; per lane two contiguous 8-half runs -> b128 pairs.
__device__ __forceinline__ v16h load_Afrag_vec(const half_t* A, int stride, int lane) {
  const int m  = lane & 15;
  const int ko = (lane & 16) ? 8 : 0;     // K offset for upper half-wave
  const half_t* p = A + m * stride + ko;
  v8h lo = *(const v8h*)(p);              // K = ko .. ko+7
  v8h hi = *(const v8h*)(p + 16);         // K = 16+ko .. 16+ko+7
  return cat8(lo, hi);
}

// A (16x32) where logical A[m][k] lives at Akm[k*stride + m] (k-major, scattered).
__device__ __forceinline__ v16h load_AfragT(const half_t* Akm, int stride, int lane) {
  v16h a;
  const int m  = lane & 15;
  const int ko = (lane & 16) ? 8 : 0;
#pragma unroll
  for (int j = 0; j < 8; ++j) {
    const int kb = ((j < 4) ? (2 * j) : (16 + 2 * (j - 4))) + ko;
    a[2 * j]     = Akm[kb * stride + m];
    a[2 * j + 1] = Akm[(kb + 1) * stride + m];
  }
  return a;
}

// B (32x16) where logical B[k][n] lives at Bnk[n*stride + k] (n-major, contiguous k).
__device__ __forceinline__ v16h load_BfragN_vec(const half_t* Bnk, int stride, int lane) {
  const int n  = lane & 15;
  const int ko = (lane & 16) ? 8 : 0;
  const half_t* p = Bnk + n * stride + ko;
  v8h lo = *(const v8h*)(p);
  v8h hi = *(const v8h*)(p + 16);
  return cat8(lo, hi);
}

// ================= Pass 1: 1x1-conv GEMM (f16 WMMA) + GroupNorm stats =================
// grid: (128 pixel-row blocks, 2 M-tiles, 8 batches x {q,k}); block: 256 (8 waves).
// C tile 128(M=out ch) x 128(N=pixels of one image row), K=256 in chunks of 32.
// Output written window-gathered: dst[((b*256+win)*64+p)*256 + c], packed b128.
__global__ __launch_bounds__(256, 1) void sdp_proj_gn(
    const float* __restrict__ x_low, const float* __restrict__ x_high,
    const float* __restrict__ wq, const float* __restrict__ wk,
    half_t* __restrict__ qraw, half_t* __restrict__ kraw,
    float* __restrict__ stats)
{
  __shared__ __attribute__((aligned(16))) half_t Af[128 * 40];  // [m][k]
  __shared__ __attribute__((aligned(16))) half_t Bf[128 * 40];  // [n][k] (transposed)
  __shared__ float sgrp[16][2];

  const int t     = threadIdx.x;
  const int lane  = t & 31;
  const int wv    = t >> 5;
  const int wr    = wv >> 1;       // 0..3 (M sub-tile)
  const int wc    = wv & 1;        // 0..1 (N sub-tile)
  const int N0    = blockIdx.x * 128;   // = one image row (row = blockIdx.x)
  const int M0    = blockIdx.y * 128;
  const int b     = blockIdx.z >> 1;
  const int which = blockIdx.z & 1;

  const float* x   = which ? x_high : x_low;
  const float* wm  = which ? wk : wq;
  half_t*      dst = which ? kraw : qraw;
  float*       st  = stats + which * 512 + b * 64;

  if (t < 32) sgrp[t >> 1][t & 1] = 0.f;
  __syncthreads();

  v8f acc[2][4];
#pragma unroll
  for (int i = 0; i < 2; ++i)
#pragma unroll
    for (int j = 0; j < 4; ++j) acc[i][j] = zero_v8f();

  const float* xb = x + (size_t)b * DIMC * HW;

  for (int kk = 0; kk < DIMC; kk += 32) {
    // stage A (weights) 128x32 -> f16, float4 reads, b64 LDS stores
#pragma unroll
    for (int it = 0; it < 4; ++it) {
      int v  = t + it * 256;
      int m  = v >> 3, k4 = (v & 7) * 4;
      v4f w4 = *(const v4f*)&wm[(M0 + m) * DIMC + kk + k4];
      v4h h4;
#pragma unroll
      for (int e = 0; e < 4; ++e) h4[e] = (half_t)w4[e];
      *(v4h*)&Af[m * 40 + k4] = h4;
    }
    // stage B (activations) 32x128 -> f16 transposed to [n][k]
#pragma unroll
    for (int it = 0; it < 4; ++it) {
      int v  = t + it * 256;
      int k  = v >> 5, n0 = (v & 31) * 4;
      v4f x4 = *(const v4f*)&xb[(size_t)(kk + k) * HW + N0 + n0];
#pragma unroll
      for (int e = 0; e < 4; ++e) Bf[(n0 + e) * 40 + k] = (half_t)x4[e];
    }
    __syncthreads();

    v16h bf4[4];
#pragma unroll
    for (int tn = 0; tn < 4; ++tn)
      bf4[tn] = load_BfragN_vec(Bf + (wc * 64 + tn * 16) * 40, 40, lane);
#pragma unroll
    for (int tm = 0; tm < 2; ++tm) {
      v16h a = load_Afrag_vec(Af + (wr * 32 + tm * 16) * 40, 40, lane);
#pragma unroll
      for (int tn = 0; tn < 4; ++tn)
        acc[tm][tn] = wmma_f16(a, bf4[tn], acc[tm][tn]);
    }
    __syncthreads();
  }

  // packed b128 stores in window-gathered layout + group stats
  const int hi_ = (lane & 16) ? 8 : 0;
  const int nl  = lane & 15;
  const int winrow = (blockIdx.x >> 3) * 16;   // window row * 16
  const int prow   = (blockIdx.x & 7) * 8;     // p row offset
#pragma unroll
  for (int tm = 0; tm < 2; ++tm) {
    float s = 0.f, sq = 0.f;
    const int mbase = M0 + wr * 32 + tm * 16 + hi_;   // 8-aligned -> one group
#pragma unroll
    for (int tn = 0; tn < 4; ++tn) {
      const int col = wc * 64 + tn * 16 + nl;
      const int win = winrow + (col >> 3);
      const int p   = prow + (col & 7);
      v8h o;
#pragma unroll
      for (int r = 0; r < 8; ++r) {
        float v = acc[tm][tn][r];
        s += v; sq += v * v;
        o[r] = (half_t)v;
      }
      *(v8h*)&dst[(((size_t)(b * 256 + win)) * 64 + p) * DIMC + mbase] = o;
    }
    const int lg = (wr * 32 + tm * 16 + hi_) >> 3;
    atomicAdd(&sgrp[lg][0], s);
    atomicAdd(&sgrp[lg][1], sq);
  }
  __syncthreads();
  if (t < 16) {
    atomicAdd(&st[((M0 >> 3) + t) * 2 + 0], sgrp[t][0]);
    atomicAdd(&st[((M0 >> 3) + t) * 2 + 1], sgrp[t][1]);
  }
}

// ================= Pass 2: per-window attention (f16 WMMA) + residual =================
// grid: 2048 windows; block: 128 (4 waves). k window DMA'd to LDS asynchronously.
__global__ __launch_bounds__(128, 1) void sdp_win_attn(
    const half_t* __restrict__ qraw, const half_t* __restrict__ kraw,
    const float* __restrict__ stats,
    const float* __restrict__ gq, const float* __restrict__ bq,
    const float* __restrict__ gk, const float* __restrict__ bk,
    const float* __restrict__ x_low, float* __restrict__ out)
{
  __shared__ __attribute__((aligned(16))) half_t knS[64 * 264];   // [p][c] (33,792 B)
  __shared__ float  S_T[64 * 65];                                 // (16,640 B)
  __shared__ __attribute__((aligned(16))) half_t attnL[64 * 72];  // [p][p'] ( 9,216 B)
  __shared__ float  qsc[256], qsh[256], ksc[256], ksh[256];       // ( 4,096 B)

  const int t    = threadIdx.x;
  const int lane = t & 31;
  const int wv   = t >> 5;           // 0..3
  const int widx = blockIdx.x;
  const int b    = widx >> 8;
  const int rem  = widx & 255;
  const int pixbase = ((rem >> 4) * 8) * 128 + (rem & 15) * 8;

  const half_t* kwin = kraw + (size_t)widx * 64 * DIMC;  // contiguous 32 KB
  const half_t* qwin = qraw + (size_t)widx * 64 * DIMC;

  // 1) launch async DMA of the raw k window into LDS (16B chunks, padded rows)
#pragma unroll
  for (int j = 0; j < 16; ++j) {
    const int chunk = t + j * 128;          // 2048 chunks
    const int p  = chunk >> 5;
    const int c0 = (chunk & 31) * 8;
    async_copy_b128(lds_addr_of(&knS[p * 264 + c0]), kwin + p * DIMC + c0);
  }

  // 2) fold GroupNorm into per-channel scale/shift while the DMA flies
  const float cntinv = 1.f / (8.f * 16384.f);
  for (int o = t; o < 256; o += 128) {
    const int g = o >> 3;
    float mq = stats[b * 64 + g * 2 + 0] * cntinv;
    float vq = stats[b * 64 + g * 2 + 1] * cntinv - mq * mq;
    float iq = rsqrtf(vq + EPSV);
    qsc[o] = iq * gq[o];
    qsh[o] = bq[o] - mq * iq * gq[o];
    float mk = stats[512 + b * 64 + g * 2 + 0] * cntinv;
    float vk = stats[512 + b * 64 + g * 2 + 1] * cntinv - mk * mk;
    float ik = rsqrtf(vk + EPSV);
    ksc[o] = ik * gk[o];
    ksh[o] = bk[o] - mk * ik * gk[o];
  }
  wait_asynccnt0();
  __syncthreads();

  // 3) normalize k in place (b128 LDS read-modify-write)
#pragma unroll
  for (int j = 0; j < 16; ++j) {
    const int chunk = t + j * 128;
    const int p  = chunk >> 5;
    const int c0 = (chunk & 31) * 8;
    v8h v = *(v8h*)&knS[p * 264 + c0];
    v8h o;
#pragma unroll
    for (int e = 0; e < 8; ++e)
      o[e] = (half_t)((float)v[e] * ksc[c0 + e] + ksh[c0 + e]);
    *(v8h*)&knS[p * 264 + c0] = o;
  }
  __syncthreads();

  // 4) GEMM1: S^T[p'][p] = sum_c kn[p'][c] * qn[p][c]
  //    A = kn rows (LDS, contiguous b128 runs); B = qn columns straight from
  //    global (window-gathered layout -> 2x global_load_b128 + normalize).
  v8f accS[4];
#pragma unroll
  for (int i = 0; i < 4; ++i) accS[i] = zero_v8f();

  const int pq = wv * 16 + (lane & 15);         // this lane's column p
  const half_t* qrow = qwin + pq * DIMC;
  const int bko = (lane & 16) ? 8 : 0;

  for (int ch = 0; ch < 8; ++ch) {
    const int c0 = ch * 32 + bko;
    v8h lo = *(const v8h*)(qrow + c0);
    v8h hp = *(const v8h*)(qrow + c0 + 16);
    v16h bq_;
#pragma unroll
    for (int e = 0; e < 8; ++e) {
      bq_[e]     = (half_t)((float)lo[e] * qsc[c0 + e]     + qsh[c0 + e]);
      bq_[e + 8] = (half_t)((float)hp[e] * qsc[c0 + 16 + e] + qsh[c0 + 16 + e]);
    }
#pragma unroll
    for (int tm = 0; tm < 4; ++tm) {
      v16h a = load_Afrag_vec(knS + (tm * 16) * 264 + ch * 32, 264, lane);
      accS[tm] = wmma_f16(a, bq_, accS[tm]);
    }
  }

  {
    const int n   = wv * 16 + (lane & 15);
    const int hi_ = (lane & 16) ? 8 : 0;
#pragma unroll
    for (int tm = 0; tm < 4; ++tm)
#pragma unroll
      for (int r = 0; r < 8; ++r)
        S_T[(tm * 16 + hi_ + r) * 65 + n] = accS[tm][r] * 0.0625f;  // 1/sqrt(256)
  }
  __syncthreads();

  // 5) softmax over p' (rows of S^T) per column p
  if (t < 64) {
    float mx = -1e30f;
    for (int i = 0; i < 64; ++i) mx = fmaxf(mx, S_T[i * 65 + t]);
    float sum = 0.f;
    for (int i = 0; i < 64; ++i) sum += __expf(S_T[i * 65 + t] - mx);
    float inv = 1.f / sum;
    for (int i = 0; i < 64; ++i)
      attnL[t * 72 + i] = (half_t)(__expf(S_T[i * 65 + t] - mx) * inv);
  }
  __syncthreads();

  // 6) GEMM2: outT[c][p] = sum_{p'} kn[p'][c] * attn[p][p']
  v8f accO[4][4];
#pragma unroll
  for (int i = 0; i < 4; ++i)
#pragma unroll
    for (int j = 0; j < 4; ++j) accO[i][j] = zero_v8f();

  for (int ch = 0; ch < 2; ++ch) {
    v16h bfr[4];
#pragma unroll
    for (int tn = 0; tn < 4; ++tn)
      bfr[tn] = load_BfragN_vec(attnL + (tn * 16) * 72 + ch * 32, 72, lane);
#pragma unroll
    for (int tm = 0; tm < 4; ++tm) {
      v16h a = load_AfragT(knS + (ch * 32) * 264 + (wv * 64 + tm * 16), 264, lane);
#pragma unroll
      for (int tn = 0; tn < 4; ++tn)
        accO[tm][tn] = wmma_f16(a, bfr[tn], accO[tm][tn]);
    }
  }

  // 7) residual add + store to (b,c,h,w)
  const float* xb = x_low + (size_t)b * DIMC * HW;
  float*       ob = out   + (size_t)b * DIMC * HW;
  const int nl  = lane & 15;
  const int hi_ = (lane & 16) ? 8 : 0;
#pragma unroll
  for (int tm = 0; tm < 4; ++tm) {
#pragma unroll
    for (int tn = 0; tn < 4; ++tn) {
      const int p   = tn * 16 + nl;
      const int pix = pixbase + ((p >> 3) << 7) + (p & 7);
#pragma unroll
      for (int r = 0; r < 8; ++r) {
        const int c = wv * 64 + tm * 16 + hi_ + r;
        const size_t idx = (size_t)c * HW + pix;
        ob[idx] = accO[tm][tn][r] + xb[idx];
      }
    }
  }
}

extern "C" void kernel_launch(void* const* d_in, const int* in_sizes, int n_in,
                              void* d_out, int out_size, void* d_ws, size_t ws_size,
                              hipStream_t stream) {
  (void)in_sizes; (void)n_in; (void)out_size; (void)ws_size;

  const float* x_low  = (const float*)d_in[0];
  const float* x_high = (const float*)d_in[1];
  const float* wq     = (const float*)d_in[2];
  const float* wk     = (const float*)d_in[3];
  const float* gq     = (const float*)d_in[4];
  const float* bq     = (const float*)d_in[5];
  const float* gk     = (const float*)d_in[6];
  const float* bk     = (const float*)d_in[7];

  const size_t rawBytes = (size_t)8 * DIMC * HW * sizeof(half_t);  // 64 MiB each
  half_t* qraw  = (half_t*)d_ws;
  half_t* kraw  = (half_t*)((char*)d_ws + rawBytes);
  float*  stats = (float*)((char*)d_ws + 2 * rawBytes);

  hipMemsetAsync(stats, 0, 1024 * sizeof(float), stream);

  dim3 g1(128, 2, 16), b1(256, 1, 1);
  sdp_proj_gn<<<g1, b1, 0, stream>>>(x_low, x_high, wq, wk, qraw, kraw, stats);

  dim3 g2(2048, 1, 1), b2(128, 1, 1);
  sdp_win_attn<<<g2, b2, 0, stream>>>(qraw, kraw, stats, gq, bq, gk, bk,
                                      x_low, (float*)d_out);
}
